// LatentODE_41755672052458
// MI455X (gfx1250) — matrix operand order repaired and checked by reading, not compiled, not run on hardware
//
#include <hip/hip_runtime.h>
#include <hip/hip_bf16.h>
#include <math.h>
#include <stdint.h>

typedef _Float16 f16;
typedef __attribute__((ext_vector_type(16))) _Float16 v16h;
typedef __attribute__((ext_vector_type(8)))  float    v8f;

#define TT   200
#define BB   4096
#define OBS  64
#define HID  128
#define NENC 50

// ---------------- WMMA fragment helpers (CDNA5 16x16x32 f16, wave32) ----------------
__device__ __forceinline__ v16h lda_frag(const f16* A, int lda, int lane) {
  const int r = lane & 15;
  const int h8 = (lane >> 4) * 8;
  const f16* p = A + r * lda + h8;
  v16h a;
#pragma unroll
  for (int i = 0; i < 8; ++i) a[i] = p[i];
#pragma unroll
  for (int i = 0; i < 8; ++i) a[8 + i] = p[16 + i];
  return a;
}

// B fragment from swizzled weight storage: [tile][lane][16 f16 contiguous]
__device__ __forceinline__ v16h ldb_swz(const f16* base, int tile, int lane) {
  const f16* p = base + (size_t)(tile * 32 + lane) * 16;
  v16h b;
#pragma unroll
  for (int i = 0; i < 16; ++i) b[i] = p[i];
  return b;
}

__device__ __forceinline__ v8f wmma_f16(v16h a, v16h b, v8f c) {
  return __builtin_amdgcn_wmma_f32_16x16x32_f16(false, a, false, b, (short)0, c,
                                                false, false);
}

// ---------------- fast transcendentals (latency-critical recurrences) ----------------
__device__ __forceinline__ float fast_rcp(float x) {
#if __has_builtin(__builtin_amdgcn_rcpf)
  return __builtin_amdgcn_rcpf(x);
#else
  return 1.f / x;
#endif
}
__device__ __forceinline__ float fast_sigmoid(float x) {
  return fast_rcp(1.f + __expf(-x));
}
__device__ __forceinline__ float fast_tanh(float x) {
#if __has_builtin(__builtin_amdgcn_tanhf)
  return __builtin_amdgcn_tanhf(x);
#else
  float e = __expf(x + x);
  return 1.f - 2.f * fast_rcp(e + 1.f);
#endif
}

// ---------------- CDNA5 async global->LDS copy (ASYNCcnt tracked) ----------------
__device__ __forceinline__ void async_b128_to_lds(void* lds, const void* gptr) {
  unsigned loff = (unsigned)(uintptr_t)lds;  // LDS offset = low 32 bits
  asm volatile("global_load_async_to_lds_b128 %0, %1, off"
               :: "v"(loff), "v"(gptr)
               : "memory");
}
__device__ __forceinline__ void wait_async0() {
#if __has_builtin(__builtin_amdgcn_s_wait_asynccnt)
  __builtin_amdgcn_s_wait_asynccnt(0);
#else
  asm volatile("s_wait_asynccnt 0x0" ::: "memory");
#endif
}

// ---------------- weight packing: f32 -> swizzled padded f16 fragments ----------------
__device__ __forceinline__ void pack_swz(int x, f16* __restrict__ dst,
                                         const float* __restrict__ src, int NT,
                                         int ldsrc, int realK, int realN) {
  int tile = x >> 9;
  int within = x & 511;
  int lane = within >> 4, i = within & 15;
  int kt = tile / NT, nt = tile - kt * NT;
  int k = kt * 32 + ((lane >> 4) << 3) + (i & 7) + ((i >> 3) << 4);
  int n = nt * 16 + (lane & 15);
  float v = (k < realK && n < realN) ? src[n * ldsrc + k] : 0.f;
  dst[x] = (f16)v;
}

__global__ void pack_kernel(const float* __restrict__ Wih, const float* __restrict__ Whh,
                            const float* __restrict__ dW1, const float* __restrict__ dW2,
                            const float* __restrict__ dc1, const float* __restrict__ dc2,
                            const float* __restrict__ dW3,
                            f16* __restrict__ WihT, f16* __restrict__ WhhT,
                            f16* __restrict__ W1p, f16* __restrict__ W2p,
                            f16* __restrict__ D1p, f16* __restrict__ D2p,
                            f16* __restrict__ W3p) {
  const int N0 = 24576, N1 = 49152, N2 = 4096, N3 = 16384, N4 = 2048, N5 = 4096,
            N6 = 2048;
  int x = blockIdx.x * blockDim.x + threadIdx.x;
  if (x < N0) { pack_swz(x, WihT, Wih, 24, 64, 64, 384); return; }
  x -= N0;
  if (x < N1) { pack_swz(x, WhhT, Whh, 24, 128, 128, 384); return; }
  x -= N1;
  if (x < N2) { pack_swz(x, W1p, dW1, 8, 3, 3, 100); return; }
  x -= N2;
  if (x < N3) { pack_swz(x, W2p, dW2, 8, 100, 100, 100); return; }
  x -= N3;
  if (x < N4) { pack_swz(x, D1p, dc1, 4, 3, 3, 64); return; }
  x -= N4;
  if (x < N5) { pack_swz(x, D2p, dc2, 4, 64, 64, 64); return; }
  x -= N5;
  if (x < N6) { pack_swz(x, W3p, dW3, 1, 100, 100, 3); return; }
}

// ---------------- GRU encoder + fused z0 heads ----------------
// WG = 32 batch rows, 16 waves; h in registers; weights register-resident.
__global__ __launch_bounds__(512) void gru_encode_kernel(
    const float* __restrict__ obs, const f16* __restrict__ WihT,
    const f16* __restrict__ WhhT, const float* __restrict__ bih,
    const float* __restrict__ bhh, const float* __restrict__ eps,
    const float* __restrict__ meanW, const float* __restrict__ meanb,
    const float* __restrict__ lvW, const float* __restrict__ lvb,
    float* __restrict__ z0m, float* __restrict__ z0lv, float* __restrict__ z0) {
  __shared__ __align__(16) float sXstage[32 * 64];  // async landing buffer (f32)
  __shared__ __align__(16) f16 sX[32 * 64];
  __shared__ __align__(16) f16 sH[32 * 128];
  const int tid = threadIdx.x;
  const int lane = tid & 31, wave = tid >> 5;  // 16 waves
  const int rt = wave >> 3, ct = wave & 7;     // row tile 0..1, gate-col tile 0..7
  const int m0 = rt * 16, j0 = ct * 16;
  const int brow0 = blockIdx.x * 32;

  const int jc = j0 + (lane & 15);
  const float bsr = bih[jc] + bhh[jc];
  const float bsz = bih[128 + jc] + bhh[128 + jc];
  const float bin_ = bih[256 + jc];
  const float bhn_ = bhh[256 + jc];

  // hoist all 18 weight B-fragments into registers for the whole recurrence
  v16h fWih[2][3], fWhh[4][3];
#pragma unroll
  for (int kt = 0; kt < 2; ++kt)
#pragma unroll
    for (int g = 0; g < 3; ++g)
      fWih[kt][g] = ldb_swz(WihT, kt * 24 + g * 8 + ct, lane);
#pragma unroll
  for (int kt = 0; kt < 4; ++kt)
#pragma unroll
    for (int g = 0; g < 3; ++g)
      fWhh[kt][g] = ldb_swz(WhhT, kt * 24 + g * 8 + ct, lane);

  for (int i = tid; i < 32 * 128; i += 512) sH[i] = (f16)0.f;
  v8f hold = {};
  async_b128_to_lds(&sXstage[tid << 2],
                    obs + ((size_t)brow0 + (tid >> 4)) * OBS + ((tid & 15) << 2));
  __syncthreads();

  for (int t = 0; t < NENC; ++t) {
    wait_async0();
    __syncthreads();  // staged frame visible to all waves
    for (int i = tid; i < 32 * 64; i += 512) sX[i] = (f16)sXstage[i];
    __syncthreads();  // sX ready, sXstage free
    if (t + 1 < NENC)  // stream next frame behind the WMMA work below
      async_b128_to_lds(&sXstage[tid << 2],
                        obs + ((size_t)(t + 1) * BB + brow0 + (tid >> 4)) * OBS +
                            ((tid & 15) << 2));

    v8f csr, csz, cin, chn;
#pragma unroll
    for (int r = 0; r < 8; ++r) { csr[r] = bsr; csz[r] = bsz; cin[r] = bin_; chn[r] = bhn_; }

#pragma unroll
    for (int kt = 0; kt < 2; ++kt) {  // x @ Wih^T
      v16h a = lda_frag(sX + m0 * 64 + kt * 32, 64, lane);
      csr = wmma_f16(a, fWih[kt][0], csr);
      csz = wmma_f16(a, fWih[kt][1], csz);
      cin = wmma_f16(a, fWih[kt][2], cin);
    }
#pragma unroll
    for (int kt = 0; kt < 4; ++kt) {  // h @ Whh^T
      v16h a = lda_frag(sH + m0 * 128 + kt * 32, 128, lane);
      csr = wmma_f16(a, fWhh[kt][0], csr);
      csz = wmma_f16(a, fWhh[kt][1], csz);
      chn = wmma_f16(a, fWhh[kt][2], chn);
    }
    __syncthreads();  // all waves done reading sH/sX

    const int rowb = m0 + (lane >> 4) * 8;
#pragma unroll
    for (int r = 0; r < 8; ++r) {
      float rg = fast_sigmoid(csr[r]);
      float zg = fast_sigmoid(csz[r]);
      float ng = fast_tanh(cin[r] + rg * chn[r]);
      float hn = (1.f - zg) * ng + zg * hold[r];
      hold[r] = hn;
      sH[(rowb + r) * 128 + jc] = (f16)hn;
    }
    __syncthreads();
  }

  // fused z0 heads: this WG's final h is in sH
  if (tid < 96) {
    int row = tid / 3, l = tid - row * 3;
    float m = meanb[l], lv = lvb[l];
    const f16* hr = sH + row * 128;
    for (int k = 0; k < 128; ++k) {
      float hv = (float)hr[k];
      m += hv * meanW[l * 128 + k];
      lv += hv * lvW[l * 128 + k];
    }
    int gidx = (brow0 + row) * 3 + l;
    z0m[gidx] = m;
    z0lv[gidx] = lv;
    z0[gidx] = m + eps[gidx] * __expf(0.5f * lv);
  }
}

// ---------------- ODE integrate (RK4) + decode: WG = 64 batch rows, 8 waves ----------------
// Per-wave column tile is constant => all weight fragments register-resident.
__global__ __launch_bounds__(256) void ode_decode_kernel(
    const float* __restrict__ z0in, const float* __restrict__ ts,
    const f16* __restrict__ W1p, const f16* __restrict__ W2p,
    const f16* __restrict__ D1p, const f16* __restrict__ D2p,
    const f16* __restrict__ W3p,
    const float* __restrict__ b1, const float* __restrict__ b2,
    const float* __restrict__ b3,
    const float* __restrict__ db1, const float* __restrict__ db2,
    float* __restrict__ recon, float* __restrict__ traj) {
  __shared__ __align__(16) f16 sIn[64 * 32];
  __shared__ __align__(16) f16 sH1[64 * 128];
  __shared__ __align__(16) f16 sH2[64 * 128];
  __shared__ float sF[64 * 16];  // layer3 WMMA output (cols 0..2 valid)
  __shared__ float sB1[128], sB2[128], sB3[16], sDb1[64], sDb2[64];
  __shared__ float sZ0[192], sZs[192], sAcc[192];
  const int tid = threadIdx.x;
  const int lane = tid & 31, wave = tid >> 5;  // 8 waves
  const int brow0 = blockIdx.x * 64;
  f16* sD = sH1;  // decoder hidden reuses sH1

  // register-resident weight fragments (per-wave constant column tiles)
  v16h fW1 = ldb_swz(W1p, wave, lane);
  v16h fW2[4];
#pragma unroll
  for (int kt = 0; kt < 4; ++kt) fW2[kt] = ldb_swz(W2p, kt * 8 + wave, lane);
  v16h fD1 = ldb_swz(D1p, wave & 3, lane);
  v16h fD2[2];
#pragma unroll
  for (int kt = 0; kt < 2; ++kt) fD2[kt] = ldb_swz(D2p, kt * 4 + (wave & 3), lane);
  v16h fW3[4] = {};
  if (wave < 4) {
#pragma unroll
    for (int kt = 0; kt < 4; ++kt) fW3[kt] = ldb_swz(W3p, kt, lane);
  }

  if (tid < 128) { sB1[tid] = (tid < 100) ? b1[tid] : 0.f; sB2[tid] = (tid < 100) ? b2[tid] : 0.f; }
  if (tid < 64) { sDb1[tid] = db1[tid]; sDb2[tid] = db2[tid]; }
  if (tid < 16) sB3[tid] = (tid < 3) ? b3[tid] : 0.f;
  if (tid < 192) sZ0[tid] = z0in[(size_t)brow0 * 3 + tid];
  __syncthreads();

  for (int t = 0; t < TT; ++t) {
    // ---- emit traj + decode current state ----
    if (tid < 192) {
      traj[((size_t)t * BB + brow0) * 3 + tid] = sZ0[tid];
      sZs[tid] = sZ0[tid];
    }
    for (int i = tid; i < 64 * 32; i += 256) {
      int row = i >> 5, col = i & 31;
      sIn[i] = (col < 3) ? (f16)sZ0[row * 3 + col] : (f16)0.f;
    }
    __syncthreads();
    {
      const int n = (wave & 3) * 16 + (lane & 15);
#pragma unroll
      for (int tile = wave; tile < 16; tile += 8) {  // relu(z @ decW1^T), N=64
        int rtw = tile >> 2;
        v8f c;
#pragma unroll
        for (int r = 0; r < 8; ++r) c[r] = sDb1[n];
        c = wmma_f16(lda_frag(sIn + rtw * 16 * 32, 32, lane), fD1, c);
        int rowb = rtw * 16 + (lane >> 4) * 8;
#pragma unroll
        for (int r = 0; r < 8; ++r) sD[(rowb + r) * 64 + n] = (f16)fmaxf(c[r], 0.f);
      }
      __syncthreads();
#pragma unroll
      for (int tile = wave; tile < 16; tile += 8) {  // h @ decW2^T -> recon (global)
        int rtw = tile >> 2;
        v8f c;
#pragma unroll
        for (int r = 0; r < 8; ++r) c[r] = sDb2[n];
#pragma unroll
        for (int kt = 0; kt < 2; ++kt)
          c = wmma_f16(lda_frag(sD + rtw * 16 * 64 + kt * 32, 64, lane), fD2[kt], c);
        int rowb = rtw * 16 + (lane >> 4) * 8;
#pragma unroll
        for (int r = 0; r < 8; ++r)
          recon[((size_t)t * BB + brow0 + rowb + r) * 64 + n] = c[r];
      }
    }
    __syncthreads();

    if (t == TT - 1) break;
    const float dt = ts[t + 1] - ts[t];

    // ---- RK4: 4 dependent evals of the 3->100->100->3 tanh MLP ----
    for (int s = 0; s < 4; ++s) {
      for (int i = tid; i < 64 * 32; i += 256) {
        int row = i >> 5, col = i & 31;
        sIn[i] = (col < 3) ? (f16)sZs[row * 3 + col] : (f16)0.f;
      }
      __syncthreads();
      {
        const int n = wave * 16 + (lane & 15);
#pragma unroll
        for (int tile = wave; tile < 32; tile += 8) {  // layer1 (N padded to 128)
          int rtw = tile >> 3;
          v8f c;
#pragma unroll
          for (int r = 0; r < 8; ++r) c[r] = sB1[n];
          c = wmma_f16(lda_frag(sIn + rtw * 16 * 32, 32, lane), fW1, c);
          int rowb = rtw * 16 + (lane >> 4) * 8;
#pragma unroll
          for (int r = 0; r < 8; ++r) sH1[(rowb + r) * 128 + n] = (f16)fast_tanh(c[r]);
        }
        __syncthreads();
#pragma unroll
        for (int tile = wave; tile < 32; tile += 8) {  // layer2 (K=N=128 padded)
          int rtw = tile >> 3;
          v8f c;
#pragma unroll
          for (int r = 0; r < 8; ++r) c[r] = sB2[n];
#pragma unroll
          for (int kt = 0; kt < 4; ++kt)
            c = wmma_f16(lda_frag(sH1 + rtw * 16 * 128 + kt * 32, 128, lane), fW2[kt], c);
          int rowb = rtw * 16 + (lane >> 4) * 8;
#pragma unroll
          for (int r = 0; r < 8; ++r) sH2[(rowb + r) * 128 + n] = (f16)fast_tanh(c[r]);
        }
      }
      __syncthreads();
      if (wave < 4) {  // layer3 via WMMA (N padded 3->16), waves 0..3
        const int n = lane & 15;
        v8f c;
#pragma unroll
        for (int r = 0; r < 8; ++r) c[r] = sB3[n];
#pragma unroll
        for (int kt = 0; kt < 4; ++kt)
          c = wmma_f16(lda_frag(sH2 + wave * 16 * 128 + kt * 32, 128, lane), fW3[kt], c);
        int rowb = wave * 16 + (lane >> 4) * 8;
#pragma unroll
        for (int r = 0; r < 8; ++r) sF[(rowb + r) * 16 + n] = c[r];
      }
      __syncthreads();
      if (tid < 192) {  // RK4 stage combine
        int row = tid / 3, comp = tid - row * 3;
        float f = sF[row * 16 + comp];
        float z0v = sZ0[tid];
        if (s == 0)      { sAcc[tid] = f;        sZs[tid] = z0v + 0.5f * dt * f; }
        else if (s == 1) { sAcc[tid] += 2.f * f; sZs[tid] = z0v + 0.5f * dt * f; }
        else if (s == 2) { sAcc[tid] += 2.f * f; sZs[tid] = z0v + dt * f; }
        else             { sAcc[tid] += f; }
      }
      __syncthreads();
    }
    if (tid < 192) sZ0[tid] += (dt * (1.f / 6.f)) * sAcc[tid];
    __syncthreads();
  }
}

extern "C" void kernel_launch(void* const* d_in, const int* in_sizes, int n_in,
                              void* d_out, int out_size, void* d_ws, size_t ws_size,
                              hipStream_t stream) {
  (void)in_sizes; (void)n_in; (void)out_size; (void)ws_size;
  const float* obs  = (const float*)d_in[0];
  const float* eps  = (const float*)d_in[1];
  const float* Wih  = (const float*)d_in[2];
  const float* Whh  = (const float*)d_in[3];
  const float* bih  = (const float*)d_in[4];
  const float* bhh  = (const float*)d_in[5];
  const float* meanW = (const float*)d_in[6];
  const float* meanb = (const float*)d_in[7];
  const float* lvW  = (const float*)d_in[8];
  const float* lvb  = (const float*)d_in[9];
  const float* dW1  = (const float*)d_in[10];
  const float* db1  = (const float*)d_in[11];
  const float* dW2  = (const float*)d_in[12];
  const float* db2  = (const float*)d_in[13];
  const float* dW3  = (const float*)d_in[14];
  const float* db3  = (const float*)d_in[15];
  const float* dc1  = (const float*)d_in[16];
  const float* dcb1 = (const float*)d_in[17];
  const float* dc2  = (const float*)d_in[18];
  const float* dcb2 = (const float*)d_in[19];
  const float* ts   = (const float*)d_in[20];

  float* recon = (float*)d_out;
  float* z0m   = recon + (size_t)TT * BB * OBS;
  float* z0lv  = z0m + (size_t)BB * 3;
  float* traj  = z0lv + (size_t)BB * 3;

  char* w = (char*)d_ws;
  f16* WihT = (f16*)w;  w += (size_t)24576 * sizeof(f16);
  f16* WhhT = (f16*)w;  w += (size_t)49152 * sizeof(f16);
  f16* W1p  = (f16*)w;  w += (size_t)4096 * sizeof(f16);
  f16* W2p  = (f16*)w;  w += (size_t)16384 * sizeof(f16);
  f16* D1p  = (f16*)w;  w += (size_t)2048 * sizeof(f16);
  f16* D2p  = (f16*)w;  w += (size_t)4096 * sizeof(f16);
  f16* W3p  = (f16*)w;  w += (size_t)2048 * sizeof(f16);
  float* z0 = (float*)w;  w += (size_t)BB * 3 * sizeof(float);

  const int packN = 24576 + 49152 + 4096 + 16384 + 2048 + 4096 + 2048;
  pack_kernel<<<(packN + 255) / 256, 256, 0, stream>>>(
      Wih, Whh, dW1, dW2, dc1, dc2, dW3, WihT, WhhT, W1p, W2p, D1p, D2p, W3p);
  gru_encode_kernel<<<BB / 32, 512, 0, stream>>>(obs, WihT, WhhT, bih, bhh, eps,
                                                 meanW, meanb, lvW, lvb,
                                                 z0m, z0lv, z0);
  ode_decode_kernel<<<BB / 64, 256, 0, stream>>>(z0, ts, W1p, W2p, D1p, D2p, W3p,
                                                 db1, db2, db3, dcb1, dcb2,
                                                 recon, traj);
}
